// TurboQuantAttention_52896817218097
// MI455X (gfx1250) — matrix operand order repaired and checked by reading, not compiled, not run on hardware
//
#include <hip/hip_runtime.h>
#include <hip/hip_bf16.h>
#include <math.h>
#include <stdint.h>

// ---------------------------------------------------------------------------
// GQA attention block for MI455X (gfx1250).
// - All matmuls on v_wmma_f32_16x16x32_bf16 (compute-bound: ~240 GFLOP vs
//   ~11us of HBM traffic at 23.3 TB/s -> bf16 in / f32 accumulate).
// - V projection GEMM writes V^T so the flash-attention kernel can stage both
//   K and V tiles with pure global_load_async_to_lds_b128 copies (ASYNCcnt).
// ---------------------------------------------------------------------------

typedef __attribute__((ext_vector_type(16))) __bf16 v16bf;
typedef __attribute__((ext_vector_type(8)))  __bf16 v8bf;
typedef __attribute__((ext_vector_type(4)))  __bf16 v4bf;
typedef __attribute__((ext_vector_type(8)))  float  v8f;

#define DIM       4096
#define NUM_HEADS 32
#define HEAD_DIM  128
#define KV_DIM    1024
#define SEQ       2048

static __device__ __forceinline__ v16bf cat8(v8bf lo, v8bf hi) {
  return __builtin_shufflevector(lo, hi, 0,1,2,3,4,5,6,7,8,9,10,11,12,13,14,15);
}
static __device__ __forceinline__ v8bf lds8(const __bf16* p) { return *(const v8bf*)p; }
static __device__ __forceinline__ v8f zero8() {
  v8f z = {0.f,0.f,0.f,0.f,0.f,0.f,0.f,0.f};
  return z;
}
static __device__ __forceinline__ v8f wmma_bf16(v16bf a, v16bf b, v8f c) {
  // 8 args: (neg_a, A, neg_b, B, c_mod, C, reuse_a, reuse_b)
  return __builtin_amdgcn_wmma_f32_16x16x32_bf16(false, a, false, b, (short)0, c, false, false);
}

// CDNA5 async global->LDS copy (16B per lane), tracked by ASYNCcnt.
// VDST = per-lane LDS byte offset, VADDR = per-lane 64-bit global address.
static __device__ __forceinline__ void async_load_b128(unsigned lds_off, const void* g) {
  asm volatile("global_load_async_to_lds_b128 %0, %1, off"
               :: "v"(lds_off), "v"(g) : "memory");
}
static __device__ __forceinline__ void wait_async0() {
  asm volatile("s_wait_asynccnt 0x0" ::: "memory");
}

// ---------------------------------------------------------------------------
// GEMM: C[M,N] = A[M,K] * B[K,N].  A fp32 or bf16 (row-major), B fp32 row-major.
// C is fp32 or bf16; C_TRANS stores C^T (i.e. Cv[col*M + row]).
// Block tile 128x128, BK=32, 256 threads = 8 waves, wave tile 32x64 (2x4 WMMAs).
// Requires M%128==0, N%128==0, K%32==0 (true for all calls here).
// ---------------------------------------------------------------------------
template<bool A_BF16, bool C_BF16, bool C_TRANS>
__global__ __launch_bounds__(256)
void gemm_wmma(const void* __restrict__ Av, const float* __restrict__ B,
               void* __restrict__ Cv, int M, int N, int K) {
  __shared__ __bf16 As[128][40];   // [m][k], padded (80B rows -> 16B-aligned runs)
  __shared__ __bf16 Bt[128][40];   // [n][k] (B transposed so B-fragments are contiguous)

  const int t     = threadIdx.x;
  const int lane  = t & 31;
  const int wave  = t >> 5;
  const int wm    = wave & 3;        // 4 wave-rows  * 32
  const int wn    = wave >> 2;       // 2 wave-cols  * 64
  const int khalf = lane >> 4;
  const int l16   = lane & 15;
  const int m0    = blockIdx.y * 128;
  const int n0    = blockIdx.x * 128;

  v8f acc[2][4];
  #pragma unroll
  for (int i = 0; i < 2; ++i)
    #pragma unroll
    for (int j = 0; j < 4; ++j) acc[i][j] = zero8();

  for (int k0 = 0; k0 < K; k0 += 32) {
    __syncthreads();
    // ---- stage A tile (128x32), converting fp32->bf16 in flight ----
    if constexpr (A_BF16) {
      const __bf16* A = (const __bf16*)Av;
      #pragma unroll
      for (int p = 0; p < 4; ++p) {
        int idx = (p * 256 + t) * 4;
        int mm = idx >> 5, kk = idx & 31;
        v4bf v = *(const v4bf*)(A + (size_t)(m0 + mm) * K + k0 + kk);
        *(v4bf*)&As[mm][kk] = v;
      }
    } else {
      const float* A = (const float*)Av;
      #pragma unroll
      for (int p = 0; p < 4; ++p) {
        int idx = (p * 256 + t) * 4;
        int mm = idx >> 5, kk = idx & 31;
        float4 v = *(const float4*)(A + (size_t)(m0 + mm) * K + k0 + kk);
        v4bf b;
        b[0] = (__bf16)v.x; b[1] = (__bf16)v.y; b[2] = (__bf16)v.z; b[3] = (__bf16)v.w;
        *(v4bf*)&As[mm][kk] = b;
      }
    }
    // ---- stage B tile (32x128) transposed into Bt[n][k] ----
    #pragma unroll
    for (int p = 0; p < 4; ++p) {
      int idx = (p * 256 + t) * 4;
      int kk = idx >> 7, nn = idx & 127;
      float4 v = *(const float4*)(B + (size_t)(k0 + kk) * N + n0 + nn);
      Bt[nn + 0][kk] = (__bf16)v.x;
      Bt[nn + 1][kk] = (__bf16)v.y;
      Bt[nn + 2][kk] = (__bf16)v.z;
      Bt[nn + 3][kk] = (__bf16)v.w;
    }
    __syncthreads();

    // ---- build fragments, 8 WMMAs ----
    v16bf af[2], bfr[4];
    #pragma unroll
    for (int i = 0; i < 2; ++i) {
      int m = wm * 32 + i * 16 + l16;
      af[i] = cat8(lds8(&As[m][khalf * 8]), lds8(&As[m][16 + khalf * 8]));
    }
    #pragma unroll
    for (int j = 0; j < 4; ++j) {
      int n = wn * 64 + j * 16 + l16;
      bfr[j] = cat8(lds8(&Bt[n][khalf * 8]), lds8(&Bt[n][16 + khalf * 8]));
    }
    #pragma unroll
    for (int i = 0; i < 2; ++i)
      #pragma unroll
      for (int j = 0; j < 4; ++j)
        acc[i][j] = wmma_bf16(af[i], bfr[j], acc[i][j]);
  }

  // ---- epilogue: C layout lane holds col = l16, VGPR r holds row r + 8*khalf ----
  #pragma unroll
  for (int i = 0; i < 2; ++i)
    #pragma unroll
    for (int j = 0; j < 4; ++j) {
      int col = n0 + wn * 64 + j * 16 + l16;
      #pragma unroll
      for (int r = 0; r < 8; ++r) {
        int row = m0 + wm * 32 + i * 16 + r + 8 * khalf;
        float val = acc[i][j][r];
        if constexpr (C_BF16) {
          if constexpr (C_TRANS)
            ((__bf16*)Cv)[(size_t)col * M + row] = (__bf16)val;
          else
            ((__bf16*)Cv)[(size_t)row * N + col] = (__bf16)val;
        } else {
          ((float*)Cv)[(size_t)row * N + col] = val;
        }
      }
    }
}

// ---------------------------------------------------------------------------
// Flash attention (non-causal). Block = 8 waves = 128 query rows of one head;
// each wave owns 16 query rows. Loop over 32-key blocks: K ([key][hd]) and
// V^T ([hd][key]) tiles staged with global_load_async_to_lds_b128; online
// softmax with shfl_xor butterflies over the 16-lane halves (matches WMMA
// C-layout rows r / r+8 per half); P round-trips per-wave LDS to become an
// A-fragment for the P@V WMMAs.
// ---------------------------------------------------------------------------
__global__ __launch_bounds__(256)
void attn_wmma(const __bf16* __restrict__ Q, const __bf16* __restrict__ Kp,
               const __bf16* __restrict__ VTp, __bf16* __restrict__ O) {
  __shared__ __bf16 Ks[32][136];     // [key][hd], 272B rows (16B-aligned chunks)
  __shared__ __bf16 Vt[128][40];     // [hd][key], 80B rows
  __shared__ __bf16 Ps[8][16][40];   // per-wave P staging [m][key]

  const int t      = threadIdx.x;
  const int lane   = t & 31;
  const int wave   = t >> 5;
  const int khalf  = lane >> 4;
  const int l16    = lane & 15;
  const int head   = blockIdx.y;
  const int kvhead = head >> 2;                    // 32 heads / 8 kv heads
  const int qRow0  = blockIdx.x * 128 + wave * 16;

  // Q A-fragments (16 rows x 128 hd = 4 chunks of K=32), straight from global bf16
  v16bf qf[4];
  {
    const __bf16* qp = Q + (size_t)(qRow0 + l16) * DIM + head * HEAD_DIM;
    #pragma unroll
    for (int c = 0; c < 4; ++c) {
      v8bf lo = *(const v8bf*)(qp + c * 32 + khalf * 8);
      v8bf hi = *(const v8bf*)(qp + c * 32 + 16 + khalf * 8);
      qf[c] = cat8(lo, hi);
    }
  }

  v8f o[8];
  #pragma unroll
  for (int d = 0; d < 8; ++d) o[d] = zero8();
  float m_run[8], l_run[8];
  #pragma unroll
  for (int r = 0; r < 8; ++r) { m_run[r] = -__builtin_inff(); l_run[r] = 0.f; }

  // softmax(s/sqrt(128)) via exp2: kscale = (1/sqrt(128)) * log2(e)
  const float kscale = 0.088388347648318447f * 1.4426950408889634f;

  for (int kb = 0; kb < SEQ; kb += 32) {
    __syncthreads();   // previous tiles fully consumed before overwrite
    // ---- async stage K block: 32x128 bf16 = 512 x 16B chunks ----
    #pragma unroll
    for (int p = 0; p < 2; ++p) {
      int c = p * 256 + t;
      int key = c >> 4, hc = c & 15;            // 16 chunks per 128-elem key row
      async_load_b128((unsigned)(uintptr_t)&Ks[key][hc * 8],
                      Kp + (size_t)(kb + key) * KV_DIM + kvhead * HEAD_DIM + hc * 8);
    }
    // ---- async stage V^T block: 128x32 bf16 = 512 x 16B chunks ----
    #pragma unroll
    for (int p = 0; p < 2; ++p) {
      int c = p * 256 + t;
      int d = c >> 2, kc = c & 3;               // 4 chunks per 32-key d row
      async_load_b128((unsigned)(uintptr_t)&Vt[d][kc * 8],
                      VTp + (size_t)(kvhead * HEAD_DIM + d) * SEQ + kb + kc * 8);
    }
    wait_async0();
    __syncthreads();

    // ---- scores: S = Q @ K^T, two 16x16 key tiles, K-dim = 128 (4 chunks) ----
    v8f s0 = zero8(), s1 = zero8();
    #pragma unroll
    for (int c = 0; c < 4; ++c) {
      v16bf b0 = cat8(lds8(&Ks[l16][c * 32 + khalf * 8]),
                      lds8(&Ks[l16][c * 32 + 16 + khalf * 8]));
      v16bf b1 = cat8(lds8(&Ks[16 + l16][c * 32 + khalf * 8]),
                      lds8(&Ks[16 + l16][c * 32 + 16 + khalf * 8]));
      s0 = wmma_bf16(qf[c], b0, s0);
      s1 = wmma_bf16(qf[c], b1, s1);
    }

    // ---- online softmax; VGPR r holds row r (lanes 0-15) / r+8 (lanes 16-31) ----
    float alpha[8];
    #pragma unroll
    for (int r = 0; r < 8; ++r) {
      float mx = fmaxf(s0[r], s1[r]);
      #pragma unroll
      for (int msk = 1; msk <= 8; msk <<= 1) mx = fmaxf(mx, __shfl_xor(mx, msk, 32));
      float mnew = fmaxf(m_run[r], mx);
      float a  = exp2f((m_run[r] - mnew) * kscale);
      float p0 = exp2f((s0[r] - mnew) * kscale);
      float p1 = exp2f((s1[r] - mnew) * kscale);
      float rs = p0 + p1;
      #pragma unroll
      for (int msk = 1; msk <= 8; msk <<= 1) rs += __shfl_xor(rs, msk, 32);
      l_run[r] = l_run[r] * a + rs;
      m_run[r] = mnew;
      alpha[r] = a;
      s0[r] = p0; s1[r] = p1;
    }
    #pragma unroll
    for (int d = 0; d < 8; ++d)
      #pragma unroll
      for (int r = 0; r < 8; ++r) o[d][r] *= alpha[r];

    // ---- stage P (C-layout -> LDS [m][key]) then reload as A-fragment ----
    #pragma unroll
    for (int r = 0; r < 8; ++r) {
      int m = r + 8 * khalf;
      Ps[wave][m][l16]      = (__bf16)s0[r];
      Ps[wave][m][16 + l16] = (__bf16)s1[r];
    }
    v16bf pf;
    {
      const __bf16* pp = &Ps[wave][l16][0];
      pf = cat8(*(const v8bf*)(pp + khalf * 8), *(const v8bf*)(pp + 16 + khalf * 8));
    }
    // ---- O += P @ V  (8 d-tiles of 16) ----
    #pragma unroll
    for (int d = 0; d < 8; ++d) {
      int dd = d * 16 + l16;
      v16bf vf = cat8(lds8(&Vt[dd][khalf * 8]), lds8(&Vt[dd][16 + khalf * 8]));
      o[d] = wmma_bf16(pf, vf, o[d]);
    }
  }

  // ---- normalize and store bf16 attn output at [row, head*128 + d] ----
  #pragma unroll
  for (int r = 0; r < 8; ++r) l_run[r] = 1.0f / l_run[r];
  #pragma unroll
  for (int d = 0; d < 8; ++d)
    #pragma unroll
    for (int r = 0; r < 8; ++r) {
      int row = qRow0 + r + 8 * khalf;
      int col = head * HEAD_DIM + d * 16 + l16;
      O[(size_t)row * DIM + col] = (__bf16)(o[d][r] * l_run[r]);
    }
}

// ---------------------------------------------------------------------------
extern "C" void kernel_launch(void* const* d_in, const int* in_sizes, int n_in,
                              void* d_out, int out_size, void* d_ws, size_t ws_size,
                              hipStream_t stream) {
  const float* query = (const float*)d_in[0];
  const float* key_  = (const float*)d_in[1];
  const float* value = (const float*)d_in[2];
  const float* Wq    = (const float*)d_in[3];
  const float* Wk    = (const float*)d_in[4];
  const float* Wv    = (const float*)d_in[5];
  const float* Wo    = (const float*)d_in[6];

  // workspace: q_bf16 (16MB) | k_bf16 (4MB) | vT_bf16 (4MB) | attn_bf16 (16MB)
  char* ws = (char*)d_ws;
  __bf16* qb  = (__bf16*)ws;
  __bf16* kb  = (__bf16*)(ws + (size_t)SEQ * DIM * 2);
  __bf16* vbT = (__bf16*)(ws + (size_t)SEQ * DIM * 2 + (size_t)SEQ * KV_DIM * 2);
  __bf16* ab  = (__bf16*)(ws + (size_t)SEQ * DIM * 2 + 2 * (size_t)SEQ * KV_DIM * 2);

  dim3 blk(256);
  gemm_wmma<false, true,  false><<<dim3(DIM / 128,    SEQ / 128), blk, 0, stream>>>(query, Wq, qb,  SEQ, DIM,    DIM);
  gemm_wmma<false, true,  false><<<dim3(KV_DIM / 128, SEQ / 128), blk, 0, stream>>>(key_,  Wk, kb,  SEQ, KV_DIM, DIM);
  gemm_wmma<false, true,  true ><<<dim3(KV_DIM / 128, SEQ / 128), blk, 0, stream>>>(value, Wv, vbT, SEQ, KV_DIM, DIM);
  attn_wmma<<<dim3(SEQ / 128, NUM_HEADS), blk, 0, stream>>>(qb, kb, vbT, ab);
  gemm_wmma<true,  false, false><<<dim3(DIM / 128,    SEQ / 128), blk, 0, stream>>>(ab, Wo, d_out, SEQ, DIM, DIM);
}
